// VQ_VAE_17136919511059
// MI455X (gfx1250) — matrix operand order repaired and checked by reading, not compiled, not run on hardware
//
#include <hip/hip_runtime.h>

// ---------------------------------------------------------------------------
// VQ-VAE forward for MI455X (gfx1250, wave32, WMMA).
// GEMMs: f32 global -> (float4 loads) -> RNE-packed bf16 in LDS ->
// v_wmma_f32_16x16x32_bf16, f32 accumulate. Block tile 128x64, K-tile 64.
// ---------------------------------------------------------------------------

typedef __attribute__((ext_vector_type(16))) __bf16       v16bf;
typedef __attribute__((ext_vector_type(8)))  float        v8f;
typedef __attribute__((ext_vector_type(8)))  unsigned int v8u;
typedef __attribute__((ext_vector_type(4)))  float        f4;
typedef __attribute__((ext_vector_type(2)))  float        f2;
typedef __attribute__((ext_vector_type(2)))  unsigned int u2;

// LDS row strides (bf16 elements), padded to dodge bank conflicts
#define SA 72
#define SB 72
#define BK 64

__device__ __forceinline__ unsigned int bfround(float f) {
  unsigned int u = __builtin_bit_cast(unsigned int, f);
  return (u + 0x7FFFu + ((u >> 16) & 1u)) >> 16;   // round-to-nearest-even
}
__device__ __forceinline__ unsigned int pack2bf(float lo, float hi) {
  return bfround(lo) | (bfround(hi) << 16);
}

// A fragment (16x32 bf16): lane m-row = lane&15, K-half = lane>>4,
// VGPR v holds K pair at ks*32 + (v>>2)*16 + (lane>>4)*8 + (v&3)*2.
__device__ __forceinline__ v16bf load_frag_a(const unsigned short* As, int rowBase,
                                             int lane, int ks) {
  const int mrow  = lane & 15;
  const int khalf = lane >> 4;
  const unsigned int* p = (const unsigned int*)(As + (rowBase + mrow) * SA);
  v8u t;
#pragma unroll
  for (int v = 0; v < 8; ++v) {
    const int kidx = (ks << 5) + ((v >> 2) << 4) + (khalf << 3) + ((v & 3) << 1);
    t[v] = p[kidx >> 1];
  }
  return __builtin_bit_cast(v16bf, t);
}

// B fragment (32x16 bf16), staged transposed in LDS as [n][k]: lane column =
// lane&15, VGPR v holds K pair at ks*32 + (v>>2)*16 + (lane>>4)*8 + (v&3)*2.
__device__ __forceinline__ v16bf load_frag_b(const unsigned short* Bs, int colBase,
                                             int lane, int ks) {
  const int ncol  = lane & 15;
  const int khalf = lane >> 4;
  const unsigned int* p = (const unsigned int*)(Bs + (colBase + ncol) * SB);
  v8u t;
#pragma unroll
  for (int v = 0; v < 8; ++v) {
    const int kidx = (ks << 5) + ((v >> 2) << 4) + (khalf << 3) + ((v & 3) << 1);
    t[v] = p[kidx >> 1];
  }
  return __builtin_bit_cast(v16bf, t);
}

// C = act(A[M,K] @ B + bias). BT=0: B is [K,N] row-major. BT=1: B is [N,K]
// row-major (multiply by B^T). 256 threads = 8 waves (4x2), each wave owns a
// 32x32 patch -> 2x2 WMMA tiles x 2 K-substeps = 8 WMMA per K-tile.
// Requires M % 128 == 0 and K % 4 == 0 (K remainder vs 64 is zero-padded).
template <bool BT, bool TANH, bool BIAS>
__global__ __launch_bounds__(256)
void gemm_bf16_wmma(const float* __restrict__ A, const float* __restrict__ B,
                    const float* __restrict__ bias, float* __restrict__ C,
                    int M, int N, int K) {
  __shared__ unsigned short As[128 * SA];
  __shared__ unsigned short Bs[64 * SB];

  const int tid  = threadIdx.x;
  const int lane = tid & 31;
  const int wid  = tid >> 5;
  const int wm   = wid & 3;   // 0..3 -> 32-row patch
  const int wn   = wid >> 2;  // 0..1 -> 32-col patch
  const int row0 = blockIdx.y * 128;
  const int col0 = blockIdx.x * 64;

  v8f acc[2][2] = {};

  const int ksteps = (K + BK - 1) >> 6;
  for (int kt = 0; kt < ksteps; ++kt) {
    const int k0 = kt << 6;

    // prefetch a slice of the next A tile (global_prefetch_b8 path)
    if (kt + 1 < ksteps) {
      const int pr = tid >> 4, pk = (tid & 15) << 2;
      if (k0 + BK + pk < K)
        __builtin_prefetch(A + (size_t)(row0 + pr) * K + (k0 + BK + pk), 0, 1);
    }

    // ---- stage A tile (128 x 64): float4 loads, packed bf16 b64 stores ----
#pragma unroll
    for (int i = 0; i < 8; ++i) {
      const int q  = tid + i * 256;      // quad index: 128 rows x 16 quads
      const int r  = q >> 4;
      const int kq = (q & 15) << 2;
      f4 v = {0.0f, 0.0f, 0.0f, 0.0f};
      if (k0 + kq < K)                    // K % 4 == 0 -> quad never straddles
        v = *(const f4*)(A + (size_t)(row0 + r) * K + (k0 + kq));
      u2 pk2; pk2.x = pack2bf(v.x, v.y); pk2.y = pack2bf(v.z, v.w);
      *(u2*)&As[r * SA + kq] = pk2;
    }

    // ---- stage B tile transposed (64 n x 64 k): packed bf16 b32 stores ----
    if (BT) {
#pragma unroll
      for (int i = 0; i < 8; ++i) {
        const int pidx = tid + i * 256;   // 64 n x 32 k-pairs
        const int n = pidx >> 5;
        const int k = (pidx & 31) << 1;
        const int gn = col0 + n;
        f2 v = {0.0f, 0.0f};
        if (gn < N && k0 + k < K)
          v = *(const f2*)(B + (size_t)gn * K + (k0 + k));
        *(unsigned int*)&Bs[n * SB + k] = pack2bf(v.x, v.y);
      }
    } else {
#pragma unroll
      for (int i = 0; i < 8; ++i) {
        const int pidx = tid + i * 256;   // 32 k-pairs x 64 n (n fastest)
        const int k = (pidx >> 6) << 1;
        const int n = pidx & 63;
        const int gn = col0 + n;
        float v0 = 0.0f, v1 = 0.0f;
        if (gn < N && k0 + k < K) {
          v0 = B[(size_t)(k0 + k) * N + gn];
          v1 = B[(size_t)(k0 + k + 1) * N + gn];
        }
        *(unsigned int*)&Bs[n * SB + k] = pack2bf(v0, v1);
      }
    }
    __syncthreads();

#pragma unroll
    for (int ks = 0; ks < 2; ++ks) {
      v16bf af0 = load_frag_a(As, wm * 32,      lane, ks);
      v16bf af1 = load_frag_a(As, wm * 32 + 16, lane, ks);
      v16bf bf0 = load_frag_b(Bs, wn * 32,      lane, ks);
      v16bf bf1 = load_frag_b(Bs, wn * 32 + 16, lane, ks);
      acc[0][0] = __builtin_amdgcn_wmma_f32_16x16x32_bf16(false, af0, false, bf0,
                                                          (short)0, acc[0][0], false, false);
      acc[0][1] = __builtin_amdgcn_wmma_f32_16x16x32_bf16(false, af0, false, bf1,
                                                          (short)0, acc[0][1], false, false);
      acc[1][0] = __builtin_amdgcn_wmma_f32_16x16x32_bf16(false, af1, false, bf0,
                                                          (short)0, acc[1][0], false, false);
      acc[1][1] = __builtin_amdgcn_wmma_f32_16x16x32_bf16(false, af1, false, bf1,
                                                          (short)0, acc[1][1], false, false);
    }
    __syncthreads();
  }

  // ---- epilogue: C/D layout: VGPR v -> M = v + 8*(lane>>4), N = lane&15 ----
#pragma unroll
  for (int i = 0; i < 2; ++i) {
    const int mbase = row0 + wm * 32 + i * 16 + ((lane >> 4) << 3);
#pragma unroll
    for (int j = 0; j < 2; ++j) {
      const int n = col0 + wn * 32 + j * 16 + (lane & 15);
      if (n < N) {
        const float bv = BIAS ? bias[n] : 0.0f;
#pragma unroll
        for (int v = 0; v < 8; ++v) {
          float val = acc[i][j][v] + bv;
          if (TANH) val = tanhf(val);
          C[(size_t)(mbase + v) * N + n] = val;
        }
      }
    }
  }
}

// ---------------- VQ helper kernels (deterministic, no float atomics) -------

__global__ __launch_bounds__(256)
void cb_norms_kernel(const float* __restrict__ cb, float* __restrict__ cn,
                     int Kc, int D) {
  const int j = blockIdx.x * 256 + threadIdx.x;
  if (j < Kc) {
    const float* c = cb + (size_t)j * D;
    float s = 0.0f;
    for (int d = 0; d < D; d += 4) {
      const f4 v = *(const f4*)(c + d);
      s += v.x * v.x + v.y * v.y + v.z * v.z + v.w * v.w;
    }
    cn[j] = s;
  }
}

// idx[b] = argmin_j (|c_j|^2 - 2 * S[b][j]); first index wins ties (jnp.argmin)
__global__ __launch_bounds__(256)
void argmin_kernel(const float* __restrict__ S, const float* __restrict__ cn,
                   int* __restrict__ idx, int Ncodes) {
  __shared__ float sv[256];
  __shared__ int   si[256];
  const int row = blockIdx.x, tid = threadIdx.x;
  const float* Srow = S + (size_t)row * Ncodes;
  float best = 3.402823466e38f;
  int   bi   = 0x7FFFFFFF;
  for (int j = tid; j < Ncodes; j += 256) {
    const float v = cn[j] - 2.0f * Srow[j];
    if (v < best) { best = v; bi = j; }   // strict < keeps earliest j per thread
  }
  sv[tid] = best; si[tid] = bi;
  __syncthreads();
  for (int s = 128; s > 0; s >>= 1) {
    if (tid < s) {
      const float ov = sv[tid + s]; const int oi = si[tid + s];
      if (ov < sv[tid] || (ov == sv[tid] && oi < si[tid])) { sv[tid] = ov; si[tid] = oi; }
    }
    __syncthreads();
  }
  if (tid == 0) idx[row] = si[0];
}

// quantized_st = codebook[idx] -> written to d_out (qst) AND to an aligned
// workspace copy (qa) used as the decoder GEMM input; partial[b] = sum (q-z)^2
__global__ __launch_bounds__(256)
void gather_loss_kernel(const float* __restrict__ cb, const float* __restrict__ z3,
                        const int* __restrict__ idx, float* __restrict__ qst,
                        float* __restrict__ qa, float* __restrict__ partial, int D) {
  __shared__ float sv[256];
  const int row = blockIdx.x, tid = threadIdx.x;
  const int j = idx[row];
  const float* c = cb + (size_t)j * D;
  const float* z = z3 + (size_t)row * D;
  float acc = 0.0f;
  for (int d = tid; d < D; d += 256) {
    const float cv = c[d], zv = z[d];
    qst[(size_t)row * D + d] = cv;
    qa [(size_t)row * D + d] = cv;
    const float df = cv - zv;
    acc += df * df;
  }
  sv[tid] = acc; __syncthreads();
  for (int s = 128; s > 0; s >>= 1) { if (tid < s) sv[tid] += sv[tid + s]; __syncthreads(); }
  if (tid == 0) partial[row] = sv[0];
}

__global__ __launch_bounds__(256)
void loss_final_kernel(const float* __restrict__ partial, float* __restrict__ out,
                       int n, float scale) {
  __shared__ float sv[256];
  const int tid = threadIdx.x;
  float acc = 0.0f;
  for (int i = tid; i < n; i += 256) acc += partial[i];
  sv[tid] = acc; __syncthreads();
  for (int s = 128; s > 0; s >>= 1) { if (tid < s) sv[tid] += sv[tid + s]; __syncthreads(); }
  if (tid == 0) out[0] = sv[0] * scale;
}

// ---------------------------------------------------------------------------

extern "C" void kernel_launch(void* const* d_in, const int* in_sizes, int n_in,
                              void* d_out, int out_size, void* d_ws, size_t ws_size,
                              hipStream_t stream) {
  (void)in_sizes; (void)n_in; (void)out_size; (void)ws_size;

  constexpr int Bm = 4096, G = 20000, D1 = 2048, D2 = 1024, D3 = 512, Kc = 8192;

  const float* x  = (const float*)d_in[0];
  const float* W1 = (const float*)d_in[1];  const float* b1 = (const float*)d_in[2];
  const float* W2 = (const float*)d_in[3];  const float* b2 = (const float*)d_in[4];
  const float* W3 = (const float*)d_in[5];  const float* b3 = (const float*)d_in[6];
  const float* cb = (const float*)d_in[7];
  const float* W4 = (const float*)d_in[8];  const float* b4 = (const float*)d_in[9];
  const float* W5 = (const float*)d_in[10]; const float* b5 = (const float*)d_in[11];
  const float* W6 = (const float*)d_in[12]; const float* b6 = (const float*)d_in[13];

  float* out     = (float*)d_out;
  float* loss    = out;                              // [1]
  float* x_recon = out + 1;                          // [4096*20000]
  float* qst     = out + 1 + (size_t)Bm * G;         // [4096*512] (4B aligned)

  // workspace arena (floats; every offset is a multiple of 4 floats -> 16B)
  float* ws      = (float*)d_ws;
  float* z3      = ws;                               // 4096*512
  float* cn      = z3 + (size_t)Bm * D3;             // 8192
  float* partial = cn + Kc;                          // 4096
  int*   idx     = (int*)(partial + Bm);             // 4096
  float* z1      = (float*)(idx + Bm);               // 4096*2048 (reused as h5)
  float* z2      = z1 + (size_t)Bm * D1;             // 4096*1024 (reused as h4)
  float* qa      = z2 + (size_t)Bm * D2;             // 4096*512  (aligned q copy)
  float* S       = qa + (size_t)Bm * D3;             // 4096*8192
  float* h4      = z2;
  float* h5      = z1;

  const dim3 blk(256);
  const int  gy = Bm / 128;  // 32

  // encoder
  gemm_bf16_wmma<false, true, true><<<dim3(D1 / 64, gy), blk, 0, stream>>>(x,  W1, b1, z1, Bm, D1, G);
  gemm_bf16_wmma<false, true, true><<<dim3(D2 / 64, gy), blk, 0, stream>>>(z1, W2, b2, z2, Bm, D2, D1);
  gemm_bf16_wmma<false, true, true><<<dim3(D3 / 64, gy), blk, 0, stream>>>(z2, W3, b3, z3, Bm, D3, D2);

  // vector quantization: S = z3 @ codebook^T, then argmin(|c|^2 - 2S)
  cb_norms_kernel<<<Kc / 256, blk, 0, stream>>>(cb, cn, Kc, D3);
  gemm_bf16_wmma<true, false, false><<<dim3(Kc / 64, gy), blk, 0, stream>>>(z3, cb, nullptr, S, Bm, Kc, D3);
  argmin_kernel<<<Bm, blk, 0, stream>>>(S, cn, idx, Kc);

  // forward values: quantized_st == codebook[idx]; loss == 1.25*mean((q-z)^2)
  gather_loss_kernel<<<Bm, blk, 0, stream>>>(cb, z3, idx, qst, qa, partial, D3);
  loss_final_kernel<<<1, blk, 0, stream>>>(partial, loss, Bm,
                                           1.25f / (float)((size_t)Bm * D3));

  // decoder (qa is the 16B-aligned copy of quantized_st)
  gemm_bf16_wmma<false, true,  true><<<dim3(D2 / 64, gy), blk, 0, stream>>>(qa, W4, b4, h4, Bm, D2, D3);
  gemm_bf16_wmma<false, true,  true><<<dim3(D1 / 64, gy), blk, 0, stream>>>(h4, W5, b5, h5, Bm, D1, D2);
  gemm_bf16_wmma<false, false, true><<<dim3((G + 63) / 64, gy), blk, 0, stream>>>(h5, W6, b6, x_recon, Bm, G, D1);
}
